// RoPEMultiheadAttention_71837622993153
// MI455X (gfx1250) — matrix operand order repaired and checked by reading, not compiled, not run on hardware
//
#include <hip/hip_runtime.h>

// ---------------------------------------------------------------------------
// RoPE Multihead Attention forward for MI455X (gfx1250, wave32, WMMA + TDM)
// Stages: cast/transpose -> fused QKV-GEMM+RoPE -> flash attention -> out proj
// GEMMs: v_wmma_f32_16x16x32_bf16, weight tiles staged LDS via Tensor Data
// Mover (tensor_load_to_lds, TENSORcnt), double buffered.
// ---------------------------------------------------------------------------

typedef __attribute__((ext_vector_type(16))) __bf16       v16bf;
typedef __attribute__((ext_vector_type(8)))  __bf16       v8bf;
typedef __attribute__((ext_vector_type(8)))  float        v8f;
typedef __attribute__((ext_vector_type(4)))  unsigned int u32x4;
typedef __attribute__((ext_vector_type(8)))  int          i32x8;
typedef __attribute__((ext_vector_type(4)))  int          i32x4;

#define DEV __device__ __forceinline__

#if defined(__has_builtin)
#if __has_builtin(__builtin_amdgcn_tensor_load_to_lds)
#define USE_TDM 1
#endif
#endif
#ifndef USE_TDM
#define USE_TDM 0
#endif

static constexpr int BATCH = 2;
static constexpr int SEQ   = 2048;
static constexpr int EMB   = 1024;
static constexpr int HEADS = 16;
static constexpr int HDIM  = 64;
static constexpr int E3    = 3 * EMB;     // 3072
static constexpr int MROWS = BATCH * SEQ; // 4096

DEV v8f v8f_zero() {
    v8f c;
#pragma unroll
    for (int i = 0; i < 8; ++i) c[i] = 0.0f;
    return c;
}

DEV v8f wmma_bf16(v16bf a, v16bf b, v8f c) {
    return __builtin_amdgcn_wmma_f32_16x16x32_bf16(false, a, false, b,
                                                   (short)0, c, false, false);
}

DEV void wait_tensorcnt0() {
#if defined(__has_builtin)
#if __has_builtin(__builtin_amdgcn_s_wait_tensorcnt)
    __builtin_amdgcn_s_wait_tensorcnt(0);
    return;
#endif
#endif
    asm volatile("s_wait_tensorcnt 0x0" ::: "memory");
}
DEV void wait_tensorcnt1() {
#if defined(__has_builtin)
#if __has_builtin(__builtin_amdgcn_s_wait_tensorcnt)
    __builtin_amdgcn_s_wait_tensorcnt(1);
    return;
#endif
#endif
    asm volatile("s_wait_tensorcnt 0x1" ::: "memory");
}

#if USE_TDM
// Issue a 2D TDM load: tile (tile_d0 x tile_d1) elements (2B each) from a
// row-major tensor with stride d0_stride, into LDS byte offset lds_off.
// D# bitfields per CDNA5 ISA.
DEV void tdm_load_2d_bf16(unsigned lds_off, const void* gaddr,
                          int tensor_d0, int tensor_d1,
                          int tile_d0, int tile_d1, int d0_stride) {
    unsigned long long ga = (unsigned long long)(uintptr_t)gaddr;
    u32x4 g0;
    g0[0] = 1u;                                     // count=1, user mode
    g0[1] = lds_off;                                // lds_addr (bytes)
    g0[2] = (unsigned)(ga & 0xffffffffu);           // global_addr[31:0]
    g0[3] = (unsigned)((ga >> 32) & 0x01ffffffu)    // global_addr[56:32]
          | (2u << 30);                             // type = 2 ("image")
    i32x8 g1;
    g1[0] = (int)(1u << 16);                        // data_size=1 (2 bytes)
    g1[1] = (int)(((unsigned)tensor_d0 & 0xffffu) << 16);         // dim0 lo16
    g1[2] = (int)((((unsigned)tensor_d0 >> 16) & 0xffffu)
          |       (((unsigned)tensor_d1 & 0xffffu) << 16));       // dim0 hi/dim1 lo
    g1[3] = (int)((((unsigned)tensor_d1 >> 16) & 0xffffu)
          |       (((unsigned)tile_d0 & 0xffffu) << 16));         // dim1 hi/tile0
    g1[4] = (int)((unsigned)tile_d1 & 0xffffu);                   // tile1, tile2=0
    g1[5] = d0_stride;                              // tensor_dim0_stride lo32
    g1[6] = 0;                                      // stride hi + dim1_stride lo
    g1[7] = 0;
    i32x4 z4;
    z4[0] = z4[1] = z4[2] = z4[3] = 0;
#if __clang_major__ >= 23
    i32x8 z8;
#pragma unroll
    for (int i = 0; i < 8; ++i) z8[i] = 0;
    __builtin_amdgcn_tensor_load_to_lds(g0, g1, z4, z4, z8, 0);
#else
    __builtin_amdgcn_tensor_load_to_lds(g0, g1, z4, z4, 0);
#endif
}
#endif // USE_TDM

// A operand (16x32 bf16, M x K). Lane l: M = l&15; per lane two contiguous
// 8-elt (16B) chunks: [k0+hi*8, +8) and [k0+16+hi*8, +8).
DEV v16bf load_a16x32(const __bf16* base, int stride, int m0, int k0, int lane) {
    int m  = m0 + (lane & 15);
    int hi = lane >> 4;
    const __bf16* p = base + (size_t)m * stride + k0 + hi * 8;
    v8bf lo = *(const v8bf*)p;
    v8bf hh = *(const v8bf*)(p + 16);
    v16bf a;
#pragma unroll
    for (int i = 0; i < 8; ++i) { a[i] = lo[i]; a[i + 8] = hh[i]; }
    return a;
}

// B operand (32x16 bf16, K x N) from global: lane l holds column N = n0 + (l&15),
// K = [k0+hi*8, +8) and [k0+16+hi*8, +8).
DEV v16bf load_b32x16(const __bf16* base, int stride, int k0, int n0, int lane) {
    int hi = lane >> 4;
    const __bf16* p = base + (size_t)(k0 + hi * 8) * stride + n0 + (lane & 15);
    v16bf b;
#pragma unroll
    for (int i = 0; i < 8; ++i) { b[i] = p[(size_t)i * stride]; b[i + 8] = p[(size_t)(16 + i) * stride]; }
    return b;
}

// B operand from an LDS-staged 32x64 tile (row-major, 64 elems per K row).
DEV v16bf load_b_lds(const __bf16* lb, int nsub, int lane) {
    int hi = lane >> 4;
    const __bf16* p = lb + hi * 8 * 64 + nsub + (lane & 15);
    v16bf b;
#pragma unroll
    for (int i = 0; i < 8; ++i) { b[i] = p[i * 64]; b[i + 8] = p[(16 + i) * 64]; }
    return b;
}

// ---------------------------------------------------------------------------
// Prep kernels
// ---------------------------------------------------------------------------
__global__ void cast_bf16_kernel(const float* __restrict__ s,
                                 __bf16* __restrict__ d, int n) {
    int i = blockIdx.x * blockDim.x + threadIdx.x;
    if (i < n) d[i] = (__bf16)s[i];
}

// src[rows][cols] fp32 -> dst[cols][rows] bf16 (K-major weights for B operand)
__global__ void transpose_cast_kernel(const float* __restrict__ s,
                                      __bf16* __restrict__ d,
                                      int rows, int cols) {
    int i = blockIdx.x * blockDim.x + threadIdx.x;
    if (i < rows * cols) {
        int r = i / cols, c = i - r * cols;
        d[(size_t)c * rows + r] = (__bf16)s[i];
    }
}

// ---------------------------------------------------------------------------
// Fused QKV projection + bias + RoPE.
// Block = 8 waves = 128(M) x 64(N) tile; wave = 16x64. The 32x64 bf16 weight
// tile is TDM-DMA'd into LDS, double buffered: wave0 issues tile t+1 while
// all 8 waves compute tile t from LDS.
// ---------------------------------------------------------------------------
__global__ __launch_bounds__(256) void qkv_rope_kernel(
    const __bf16* __restrict__ X,   // [4096][1024] bf16
    const __bf16* __restrict__ Wt,  // [1024][3072] bf16 (W_qkv^T)
    const float*  __restrict__ bias,// [3072]
    __bf16* __restrict__ Qb,        // [B*H][S][D]
    __bf16* __restrict__ Kb,        // [B*H][D][S]
    __bf16* __restrict__ Vb)        // [B*H][S][D]
{
    extern __shared__ __bf16 bsh[];           // 2 x (32x64) bf16 = 8 KB
    int lane = threadIdx.x & 31, wv = threadIdx.x >> 5;
    int blk = blockIdx.x;                     // 1536 blocks
    int bn = blk % 48, bm = blk / 48;         // 48 N-tiles x 32 M-blocks
    int n0 = bn * 64;
    int m0 = bm * 128 + wv * 16;

    v8f a0 = v8f_zero(), a1 = v8f_zero(), a2 = v8f_zero(), a3 = v8f_zero();

#if USE_TDM
    if (wv == 0)
        tdm_load_2d_bf16(0, Wt + n0, E3, EMB, 64, 32, E3);
    for (int it = 0; it < EMB / 32; ++it) {
        __syncthreads();                      // buf[(it+1)&1] free to overwrite
        if (wv == 0) {
            if (it + 1 < EMB / 32) {
                tdm_load_2d_bf16((unsigned)(((it + 1) & 1) * 4096),
                                 Wt + (size_t)(it + 1) * 32 * E3 + n0,
                                 E3, EMB, 64, 32, E3);
                wait_tensorcnt1();            // tile `it` landed in LDS
            } else {
                wait_tensorcnt0();
            }
        }
        __syncthreads();                      // buf[it&1] visible to block
        const __bf16* lb = bsh + (it & 1) * 2048;
        v16bf av = load_a16x32(X, EMB, m0, it * 32, lane);
        a0 = wmma_bf16(av, load_b_lds(lb, 0,  lane), a0);
        a1 = wmma_bf16(av, load_b_lds(lb, 16, lane), a1);
        a2 = wmma_bf16(av, load_b_lds(lb, 32, lane), a2);
        a3 = wmma_bf16(av, load_b_lds(lb, 48, lane), a3);
    }
#else
#pragma unroll 4
    for (int k0 = 0; k0 < EMB; k0 += 32) {
        v16bf av = load_a16x32(X, EMB, m0, k0, lane);
        a0 = wmma_bf16(av, load_b32x16(Wt, E3, k0, n0,      lane), a0);
        a1 = wmma_bf16(av, load_b32x16(Wt, E3, k0, n0 + 16, lane), a1);
        a2 = wmma_bf16(av, load_b32x16(Wt, E3, k0, n0 + 32, lane), a2);
        a3 = wmma_bf16(av, load_b32x16(Wt, E3, k0, n0 + 48, lane), a3);
    }
#endif

    int col = lane & 15, hlf = lane >> 4;
    float bv0 = bias[n0 + col],      bv1 = bias[n0 + 16 + col];
    float bv2 = bias[n0 + 32 + col], bv3 = bias[n0 + 48 + col];
#pragma unroll
    for (int e = 0; e < 8; ++e) { a0[e] += bv0; a1[e] += bv1; a2[e] += bv2; a3[e] += bv3; }

    int sec  = n0 >> 10;            // 0=q, 1=k, 2=v
    int h    = (n0 & 1023) >> 6;
    int bidx = m0 >> 11;            // batch
    int s0   = (m0 & 2047) + hlf * 8;
    int bh   = bidx * HEADS + h;

    if (sec == 2) {
#pragma unroll
        for (int e = 0; e < 8; ++e) {
            __bf16* vp = Vb + ((size_t)bh * SEQ + (s0 + e)) * HDIM;
            vp[col]      = (__bf16)a0[e];
            vp[16 + col] = (__bf16)a1[e];
            vp[32 + col] = (__bf16)a2[e];
            vp[48 + col] = (__bf16)a3[e];
        }
    } else {
        // RoPE: d = t*16 + col; rotate-half partner lives in accumulator t^2.
        float invf0 = __powf(10000.0f, -((float)col)        * (1.0f / 32.0f));
        float invf1 = __powf(10000.0f, -((float)(col + 16)) * (1.0f / 32.0f));
#pragma unroll
        for (int e = 0; e < 8; ++e) {
            int   s  = s0 + e;
            float fs = (float)s;
            float sn0, cs0, sn1, cs1;
            __sincosf(fs * invf0, &sn0, &cs0);
            __sincosf(fs * invf1, &sn1, &cs1);
            float x0 = a0[e], x1 = a1[e], x2 = a2[e], x3 = a3[e];
            float r0 = x0 * cs0 - x2 * sn0;
            float r1 = x1 * cs1 - x3 * sn1;
            float r2 = x2 * cs0 + x0 * sn0;
            float r3 = x3 * cs1 + x1 * sn1;
            if (sec == 0) {
                const float sc = 0.125f;      // 1/sqrt(64) folded into q
                __bf16* qp = Qb + ((size_t)bh * SEQ + s) * HDIM;
                qp[col]      = (__bf16)(r0 * sc);
                qp[16 + col] = (__bf16)(r1 * sc);
                qp[32 + col] = (__bf16)(r2 * sc);
                qp[48 + col] = (__bf16)(r3 * sc);
            } else {
                __bf16* kp = Kb + (size_t)bh * HDIM * SEQ + s;
                kp[(size_t)col * SEQ]        = (__bf16)r0;
                kp[(size_t)(16 + col) * SEQ] = (__bf16)r1;
                kp[(size_t)(32 + col) * SEQ] = (__bf16)r2;
                kp[(size_t)(48 + col) * SEQ] = (__bf16)r3;
            }
        }
    }
}

// ---------------------------------------------------------------------------
// Flash-style attention: one wave per 16-query tile, online softmax over
// 32-key chunks. P goes through wave-private LDS to re-layout C -> A operand.
// ---------------------------------------------------------------------------
__global__ __launch_bounds__(256) void attn_kernel(
    const __bf16* __restrict__ Q,   // [B*H][S][D], pre-scaled
    const __bf16* __restrict__ K,   // [B*H][D][S]
    const __bf16* __restrict__ V,   // [B*H][S][D]
    __bf16* __restrict__ O)         // [B][S][H*D] bf16 context
{
    __shared__ __bf16 plds[8 * 16 * 32];
    int lane = threadIdx.x & 31, wloc = threadIdx.x >> 5;
    int w  = blockIdx.x * 8 + wloc;
    int bh = w >> 7;
    int mt = w & 127;
    int m0 = mt * 16;

    const __bf16* Qh = Q + (size_t)bh * SEQ * HDIM;
    const __bf16* Kh = K + (size_t)bh * HDIM * SEQ;
    const __bf16* Vh = V + (size_t)bh * SEQ * HDIM;

    v16bf aq0 = load_a16x32(Qh, HDIM, m0, 0,  lane);
    v16bf aq1 = load_a16x32(Qh, HDIM, m0, 32, lane);

    float mrun[8], lrun[8];
#pragma unroll
    for (int e = 0; e < 8; ++e) { mrun[e] = -3.0e38f; lrun[e] = 0.0f; }
    v8f o0 = v8f_zero(), o1 = v8f_zero(), o2 = v8f_zero(), o3 = v8f_zero();
    __bf16* myp = plds + wloc * (16 * 32);

    for (int kt = 0; kt < SEQ / 32; ++kt) {
        int key0 = kt * 32;
        v8f c0 = v8f_zero(), c1 = v8f_zero();
        c0 = wmma_bf16(aq0, load_b32x16(Kh, SEQ, 0,  key0,      lane), c0);
        c0 = wmma_bf16(aq1, load_b32x16(Kh, SEQ, 32, key0,      lane), c0);
        c1 = wmma_bf16(aq0, load_b32x16(Kh, SEQ, 0,  key0 + 16, lane), c1);
        c1 = wmma_bf16(aq1, load_b32x16(Kh, SEQ, 32, key0 + 16, lane), c1);

        float corr[8];
#pragma unroll
        for (int e = 0; e < 8; ++e) {
            float rm = fmaxf(c0[e], c1[e]);
            rm = fmaxf(rm, __shfl_xor(rm, 1, 32));
            rm = fmaxf(rm, __shfl_xor(rm, 2, 32));
            rm = fmaxf(rm, __shfl_xor(rm, 4, 32));
            rm = fmaxf(rm, __shfl_xor(rm, 8, 32));
            float nm = fmaxf(mrun[e], rm);
            float cr = __expf(mrun[e] - nm);
            corr[e] = cr; mrun[e] = nm;
            float p0 = __expf(c0[e] - nm);
            float p1 = __expf(c1[e] - nm);
            float rs = p0 + p1;
            rs += __shfl_xor(rs, 1, 32);
            rs += __shfl_xor(rs, 2, 32);
            rs += __shfl_xor(rs, 4, 32);
            rs += __shfl_xor(rs, 8, 32);
            lrun[e] = lrun[e] * cr + rs;
            int row = e + (lane >> 4) * 8;
            int col = lane & 15;
            myp[row * 32 + col]      = (__bf16)p0;
            myp[row * 32 + col + 16] = (__bf16)p1;
        }
#pragma unroll
        for (int e = 0; e < 8; ++e) {
            o0[e] *= corr[e]; o1[e] *= corr[e];
            o2[e] *= corr[e]; o3[e] *= corr[e];
        }
        v16bf pa = load_a16x32(myp, 32, 0, 0, lane);
        o0 = wmma_bf16(pa, load_b32x16(Vh, HDIM, key0, 0,  lane), o0);
        o1 = wmma_bf16(pa, load_b32x16(Vh, HDIM, key0, 16, lane), o1);
        o2 = wmma_bf16(pa, load_b32x16(Vh, HDIM, key0, 32, lane), o2);
        o3 = wmma_bf16(pa, load_b32x16(Vh, HDIM, key0, 48, lane), o3);
    }

    int col = lane & 15;
    int s0  = m0 + (lane >> 4) * 8;
    int b   = bh >> 4, h = bh & 15;
#pragma unroll
    for (int e = 0; e < 8; ++e) {
        int   s   = s0 + e;
        float inv = 1.0f / lrun[e];
        __bf16* op = O + ((size_t)(b * SEQ + s) * EMB) + h * HDIM;
        op[col]      = (__bf16)(o0[e] * inv);
        op[16 + col] = (__bf16)(o1[e] * inv);
        op[32 + col] = (__bf16)(o2[e] * inv);
        op[48 + col] = (__bf16)(o3[e] * inv);
    }
}

// ---------------------------------------------------------------------------
// Output projection: out = ctx @ W_out^T + b_out (fp32), TDM-staged weights.
// ---------------------------------------------------------------------------
__global__ __launch_bounds__(256) void out_proj_kernel(
    const __bf16* __restrict__ A,    // [4096][1024] bf16 context
    const __bf16* __restrict__ Wt,   // [1024][1024] bf16 (W_out^T)
    const float*  __restrict__ bias, // [1024]
    float* __restrict__ out)         // [4096][1024] fp32
{
    extern __shared__ __bf16 bsh[];
    int lane = threadIdx.x & 31, wv = threadIdx.x >> 5;
    int blk = blockIdx.x;                     // 512 blocks
    int bn = blk & 15, bm = blk >> 4;         // 16 N-tiles x 32 M-blocks
    int n0 = bn * 64;
    int m0 = bm * 128 + wv * 16;

    v8f a0 = v8f_zero(), a1 = v8f_zero(), a2 = v8f_zero(), a3 = v8f_zero();

#if USE_TDM
    if (wv == 0)
        tdm_load_2d_bf16(0, Wt + n0, EMB, EMB, 64, 32, EMB);
    for (int it = 0; it < EMB / 32; ++it) {
        __syncthreads();
        if (wv == 0) {
            if (it + 1 < EMB / 32) {
                tdm_load_2d_bf16((unsigned)(((it + 1) & 1) * 4096),
                                 Wt + (size_t)(it + 1) * 32 * EMB + n0,
                                 EMB, EMB, 64, 32, EMB);
                wait_tensorcnt1();
            } else {
                wait_tensorcnt0();
            }
        }
        __syncthreads();
        const __bf16* lb = bsh + (it & 1) * 2048;
        v16bf av = load_a16x32(A, EMB, m0, it * 32, lane);
        a0 = wmma_bf16(av, load_b_lds(lb, 0,  lane), a0);
        a1 = wmma_bf16(av, load_b_lds(lb, 16, lane), a1);
        a2 = wmma_bf16(av, load_b_lds(lb, 32, lane), a2);
        a3 = wmma_bf16(av, load_b_lds(lb, 48, lane), a3);
    }
#else
#pragma unroll 4
    for (int k0 = 0; k0 < EMB; k0 += 32) {
        v16bf av = load_a16x32(A, EMB, m0, k0, lane);
        a0 = wmma_bf16(av, load_b32x16(Wt, EMB, k0, n0,      lane), a0);
        a1 = wmma_bf16(av, load_b32x16(Wt, EMB, k0, n0 + 16, lane), a1);
        a2 = wmma_bf16(av, load_b32x16(Wt, EMB, k0, n0 + 32, lane), a2);
        a3 = wmma_bf16(av, load_b32x16(Wt, EMB, k0, n0 + 48, lane), a3);
    }
#endif

    int col = lane & 15, hlf = lane >> 4;
    float bv0 = bias[n0 + col],      bv1 = bias[n0 + 16 + col];
    float bv2 = bias[n0 + 32 + col], bv3 = bias[n0 + 48 + col];
#pragma unroll
    for (int e = 0; e < 8; ++e) {
        int row = m0 + e + hlf * 8;
        float* op = out + (size_t)row * EMB + n0;
        op[col]      = a0[e] + bv0;
        op[16 + col] = a1[e] + bv1;
        op[32 + col] = a2[e] + bv2;
        op[48 + col] = a3[e] + bv3;
    }
}

// ---------------------------------------------------------------------------
extern "C" void kernel_launch(void* const* d_in, const int* in_sizes, int n_in,
                              void* d_out, int out_size, void* d_ws, size_t ws_size,
                              hipStream_t stream) {
    (void)in_sizes; (void)n_in; (void)out_size; (void)ws_size;
    const float* query = (const float*)d_in[0];   // [2,2048,1024]
    const float* Wqkv  = (const float*)d_in[3];   // [3072,1024]
    const float* bqkv  = (const float*)d_in[4];   // [3072]
    const float* Wout  = (const float*)d_in[5];   // [1024,1024]
    const float* bout  = (const float*)d_in[6];   // [1024]
    float* out = (float*)d_out;

    char* ws = (char*)d_ws;
    size_t o = 0;
    __bf16* Xbf   = (__bf16*)(ws + o); o += (size_t)MROWS * EMB * 2;
    __bf16* Wqkvt = (__bf16*)(ws + o); o += (size_t)EMB * E3 * 2;
    __bf16* Woutt = (__bf16*)(ws + o); o += (size_t)EMB * EMB * 2;
    __bf16* Qb    = (__bf16*)(ws + o); o += (size_t)BATCH * HEADS * SEQ * HDIM * 2;
    __bf16* Kb    = (__bf16*)(ws + o); o += (size_t)BATCH * HEADS * SEQ * HDIM * 2;
    __bf16* Vb    = (__bf16*)(ws + o); o += (size_t)BATCH * HEADS * SEQ * HDIM * 2;
    __bf16* Ob    = (__bf16*)(ws + o); o += (size_t)MROWS * EMB * 2;

    {
        int n = MROWS * EMB;
        cast_bf16_kernel<<<(n + 255) / 256, 256, 0, stream>>>(query, Xbf, n);
    }
    {
        int n = E3 * EMB;
        transpose_cast_kernel<<<(n + 255) / 256, 256, 0, stream>>>(Wqkv, Wqkvt, E3, EMB);
    }
    {
        int n = EMB * EMB;
        transpose_cast_kernel<<<(n + 255) / 256, 256, 0, stream>>>(Wout, Woutt, EMB, EMB);
    }

    // QKV GEMM + RoPE: 48 N-tiles x 32 M-blocks, 8 KB dynamic LDS (2x 32x64 bf16)
    qkv_rope_kernel<<<1536, 256, 8192, stream>>>(Xbf, Wqkvt, bqkv, Qb, Kb, Vb);

    // flash attention: 4096 wave-tiles
    attn_kernel<<<4096 / 8, 256, 0, stream>>>(Qb, Kb, Vb, Ob);

    // output projection: 16 N-tiles x 32 M-blocks
    out_proj_kernel<<<512, 256, 8192, stream>>>(Ob, Woutt, bout, out);
}
